// Einsum_89979564851975
// MI455X (gfx1250) — compile-verified, matrix operand order checked by reading, not run on hardware
//
#include <hip/hip_runtime.h>

// out[M][N] = x[M][K] @ (w[K][N] + 2.0 * lora_a[K][16] @ lora_b[16][N])
#define MDIM 8192   // B*T
#define KDIM 2048   // D
#define NDIM 8192   // F
#define RANK 16
#define SCALING 2.0f

#define BM 128
#define BN 256
#define BK 32
#define LDT 40      // padded LDS row stride (bf16 elems) = 80B -> conflict-free b128 reads
#define KTILES (KDIM / BK)

typedef __attribute__((ext_vector_type(16))) __bf16 v16bf;
typedef __attribute__((ext_vector_type(8)))  __bf16 v8bf;
typedef __attribute__((ext_vector_type(8)))  float  v8f;

static __device__ __forceinline__ unsigned bf16rne(float f) {
  unsigned u = __builtin_bit_cast(unsigned, f);
  return (u + 0x7FFFu + ((u >> 16) & 1u)) >> 16;   // round-to-nearest-even
}
static __device__ __forceinline__ unsigned bf16pack(float a, float b) {
  return bf16rne(a) | (bf16rne(b) << 16);
}

// ---- pre-pass 1: x fp32 -> bf16 (8 elements/thread) -----------------------
__global__ void cvt_x_kernel(const float4* __restrict__ x, uint4* __restrict__ xb) {
  size_t i = (size_t)blockIdx.x * blockDim.x + threadIdx.x;
  float4 v0 = x[2 * i];
  float4 v1 = x[2 * i + 1];
  uint4 o;
  o.x = bf16pack(v0.x, v0.y);
  o.y = bf16pack(v0.z, v0.w);
  o.z = bf16pack(v1.x, v1.y);
  o.w = bf16pack(v1.z, v1.w);
  xb[i] = o;
}

// ---- pre-pass 2: W_eff^T[f][d] = bf16( w[d][f] + 2 * sum_r la[d][r]*lb[r][f] )
__global__ void build_wt_kernel(const float* __restrict__ w,
                                const float* __restrict__ la,
                                const float* __restrict__ lb,
                                unsigned short* __restrict__ wt) {
  __shared__ float lbf[RANK];
  const int f = blockIdx.y;
  const int d = blockIdx.x * blockDim.x + threadIdx.x;
  if (threadIdx.x < RANK) lbf[threadIdx.x] = lb[(size_t)threadIdx.x * NDIM + f];
  __syncthreads();
  float acc = w[(size_t)d * NDIM + f];
  const float* lar = la + (size_t)d * RANK;
  float dot = 0.f;
#pragma unroll
  for (int r = 0; r < RANK; ++r) dot += lar[r] * lbf[r];
  acc += SCALING * dot;
  wt[(size_t)f * KDIM + d] = (unsigned short)bf16rne(acc);
}

// async DMA: 16B per lane, global -> LDS, tracked by ASYNCcnt
static __device__ __forceinline__ void async_b128(const unsigned short* g,
                                                  unsigned short* l) {
  unsigned lds = (unsigned)(size_t)l;   // low 32 bits of flat addr == LDS byte addr
  asm volatile("global_load_async_to_lds_b128 %0, %1, off"
               :: "v"(lds), "v"(g) : "memory");
}
static __device__ __forceinline__ void wait_async0() {
  asm volatile("s_wait_asynccnt 0x0" ::: "memory");
}

// ---- main GEMM: C[M][N] = A[M][K](bf16) * Bt[N][K](bf16), f32 accumulate ---
__launch_bounds__(256)
__global__ void gemm_bf16_wmma(const unsigned short* __restrict__ A,
                               const unsigned short* __restrict__ Bt,
                               float* __restrict__ C) {
  __shared__ __align__(16) unsigned short sA[2][BM * LDT];   // 2 x 10 KB
  __shared__ __align__(16) unsigned short sB[2][BN * LDT];   // 2 x 20 KB

  const int tid  = threadIdx.x;
  const int lane = tid & 31;
  const int wave = tid >> 5;
  const int wm   = wave >> 2;            // 0..1 : 64-row strip
  const int wn   = wave & 3;             // 0..3 : 64-col strip
  const int m0   = blockIdx.y * BM;
  const int n0   = blockIdx.x * BN;

  v8f acc[4][4];
#pragma unroll
  for (int i = 0; i < 4; ++i)
#pragma unroll
    for (int j = 0; j < 4; ++j) { v8f z = {}; acc[i][j] = z; }

  // tile staging map: thread -> (row 0..127, 16-elem K chunk)
  const int lr = tid >> 1;
  const int lc = (tid & 1) * 16;
  const unsigned short* gA  = A  + (size_t)(m0 + lr) * KDIM + lc;
  const unsigned short* gB0 = Bt + (size_t)(n0 + lr) * KDIM + lc;
  const unsigned short* gB1 = gB0 + (size_t)128 * KDIM;
  const int sa_off = lr * LDT + lc;
  const int sb0_off = lr * LDT + lc;
  const int sb1_off = (lr + 128) * LDT + lc;

  // WMMA 16-bit fragment addressing (ISA 7.12.2)
  const int krow = lane & 15;
  const int kb   = (lane >> 4) << 3;     // 0 or 8 elements

  // ---- prologue: stage tile 0 into buffer 0 ----
  async_b128(gA,      &sA[0][sa_off]);
  async_b128(gA + 8,  &sA[0][sa_off + 8]);
  async_b128(gB0,     &sB[0][sb0_off]);
  async_b128(gB0 + 8, &sB[0][sb0_off + 8]);
  async_b128(gB1,     &sB[0][sb1_off]);
  async_b128(gB1 + 8, &sB[0][sb1_off + 8]);
  wait_async0();
  __syncthreads();

  for (int kt = 0; kt < KTILES; ++kt) {
    const int cur = kt & 1;
    const int nxt = cur ^ 1;

    // stage tile kt+1 into the other buffer (overlaps with compute below)
    if (kt + 1 < KTILES) {
      const unsigned short* a  = gA  + (kt + 1) * BK;
      const unsigned short* b0 = gB0 + (kt + 1) * BK;
      const unsigned short* b1 = gB1 + (kt + 1) * BK;
      async_b128(a,      &sA[nxt][sa_off]);
      async_b128(a + 8,  &sA[nxt][sa_off + 8]);
      async_b128(b0,     &sB[nxt][sb0_off]);
      async_b128(b0 + 8, &sB[nxt][sb0_off + 8]);
      async_b128(b1,     &sB[nxt][sb1_off]);
      async_b128(b1 + 8, &sB[nxt][sb1_off + 8]);
    }

    // fragments from current buffer
    v16bf af[4], bfm[4];
#pragma unroll
    for (int tm = 0; tm < 4; ++tm) {
      const unsigned short* p = &sA[cur][(wm * 64 + tm * 16 + krow) * LDT + kb];
      v8bf lo = *(const v8bf*)(p);        // K = kb .. kb+7
      v8bf hi = *(const v8bf*)(p + 16);   // K = kb+16 .. kb+23
      af[tm] = __builtin_shufflevector(lo, hi,
               0,1,2,3,4,5,6,7,8,9,10,11,12,13,14,15);
    }
#pragma unroll
    for (int tn = 0; tn < 4; ++tn) {
      const unsigned short* p = &sB[cur][(wn * 64 + tn * 16 + krow) * LDT + kb];
      v8bf lo = *(const v8bf*)(p);
      v8bf hi = *(const v8bf*)(p + 16);
      bfm[tn] = __builtin_shufflevector(lo, hi,
               0,1,2,3,4,5,6,7,8,9,10,11,12,13,14,15);
    }

#pragma unroll
    for (int tm = 0; tm < 4; ++tm)
#pragma unroll
      for (int tn = 0; tn < 4; ++tn)
        acc[tm][tn] = __builtin_amdgcn_wmma_f32_16x16x32_bf16(
            false, af[tm], false, bfm[tn], (short)0, acc[tm][tn], false, false);

    wait_async0();      // this wave's DMA for buf[nxt] is done
    __syncthreads();    // everyone's DMA done + everyone done reading buf[cur]
  }

  // epilogue: C/D layout — VGPR i holds M=i (lanes 0-15) / M=i+8 (lanes 16-31)
  const int rhi = (lane >> 4) << 3;
  const int cn  = lane & 15;
#pragma unroll
  for (int tm = 0; tm < 4; ++tm)
#pragma unroll
    for (int tn = 0; tn < 4; ++tn) {
      const int row = m0 + wm * 64 + tm * 16 + rhi;
      const int col = n0 + wn * 64 + tn * 16 + cn;
      float* cp = C + (size_t)row * NDIM + col;
#pragma unroll
      for (int i = 0; i < 8; ++i) cp[(size_t)i * NDIM] = acc[tm][tn][i];
    }
}

extern "C" void kernel_launch(void* const* d_in, const int* in_sizes, int n_in,
                              void* d_out, int out_size, void* d_ws, size_t ws_size,
                              hipStream_t stream) {
  const float* x  = (const float*)d_in[0];   // [8192][2048]
  const float* w  = (const float*)d_in[1];   // [2048][8192]
  const float* la = (const float*)d_in[2];   // [2048][16]
  const float* lb = (const float*)d_in[3];   // [16][8192]
  float* out = (float*)d_out;                // [8192][8192]

  unsigned short* xb = (unsigned short*)d_ws;                   // 32 MB bf16 x
  unsigned short* wt = xb + (size_t)MDIM * KDIM;                // 32 MB bf16 W_eff^T
  (void)in_sizes; (void)n_in; (void)out_size; (void)ws_size;

  cvt_x_kernel<<<(MDIM * KDIM / 8) / 256, 256, 0, stream>>>(
      (const float4*)x, (uint4*)xb);

  build_wt_kernel<<<dim3(KDIM / 256, NDIM), 256, 0, stream>>>(w, la, lb, wt);

  gemm_bf16_wmma<<<dim3(NDIM / BN, MDIM / BM), 256, 0, stream>>>(xb, wt, out);
}
